// LiftSplatBEVMapper_32057635898112
// MI455X (gfx1250) — compile-verified, matrix operand order checked by reading, not run on hardware
//
#include <hip/hip_runtime.h>
#include <cstdint>
#include <cstddef>

// Problem geometry (fixed by reference)
#define NXV 256
#define NYV 256
#define BB 4
#define CC 128            // feature channels
#define HF 32
#define WF 64
#define HH 256
#define WW 512
#define NPIX (BB*HH*WW)   // 524288
#define C1 129            // conv1 input channels (128 feat + 1 depth)
#define CP 160            // padded to 5 K-chunks of 32 for bf16 WMMA
#define CO 64             // conv1 output channels
#define NG 16             // groupnorm groups (4 ch/group)
#define SVOX (BB*NXV*NYV) // 262144 voxels

#define TILE_PX 66        // 64-pixel strip + 1-pixel halo each side

#define AS1 __attribute__((address_space(1)))
#define AS3 __attribute__((address_space(3)))

typedef __attribute__((ext_vector_type(16))) __bf16 v16bf;
typedef __attribute__((ext_vector_type(8)))  float  v8f;
typedef int vsi4 __attribute__((vector_size(4 * sizeof(int))));

union Frag16 { v16bf v; uint4 q[2]; };

// K index held in half `j` (0..15) of a 16-half WMMA operand for lane-half `half`
// (per CDNA5 ISA 16-bit A/B operand layout: VGPR0..3 -> K = 8*half + 0..7,
//  VGPR4..7 -> K = 16 + 8*half + 0..7)
__device__ __forceinline__ int klocal_of(int j, int half) {
  return (j < 8) ? (8 * half + j) : (16 + 8 * half + (j - 8));
}

// ---------------------------------------------------------------- geometry ---
__global__ void geom_kernel(const float* __restrict__ depth,
                            const float* __restrict__ Km,
                            const float* __restrict__ c2e,
                            const float* __restrict__ reso,
                            int* __restrict__ idx) {
  int i = blockIdx.x * blockDim.x + threadIdx.x;
  if (i >= NPIX) return;
  int b = i / (HH * WW);
  int r = i % (HH * WW);
  int v = r / WW, u = r % WW;
  float d = depth[i];
  const float* Kb = Km + b * 9;
  float fx = Kb[0], cx = Kb[2], fy = Kb[4], cy = Kb[5];
  float xs = ((float)u - cx) * d / fx;
  float ys = ((float)v - cy) * d / fy;
  const float* T = c2e + b * 16;
  float px = T[0] * xs + T[1] * ys + T[2] * d + T[3];
  float py = T[4] * xs + T[5] * ys + T[6] * d + T[7];
  float res = reso[b];
  float ymin = -(float)NYV * res * 0.5f;
  int vx = (int)floorf(px / res);
  int vy = (int)floorf((py - ymin) / res);
  bool ok = (vx >= 0) && (vx < NXV) && (vy >= 0) && (vy < NYV);
  idx[i] = ok ? (b * (NXV * NYV) + vx * NYV + vy) : -1;
}

// ------------------------------------------- bilinear upsample + depth -> bf16
__global__ void build_hin_kernel(const float* __restrict__ x,
                                 const float* __restrict__ depth,
                                 __bf16* __restrict__ hin) {
  int i = blockIdx.x * blockDim.x + threadIdx.x;
  if (i >= NPIX) return;
  int b = i / (HH * WW);
  int r = i % (HH * WW);
  int yy = r / WW, xx = r % WW;
  // half-pixel bilinear (jax.image.resize 'bilinear'), edge-clamped
  float sy = ((float)yy + 0.5f) * ((float)HF / (float)HH) - 0.5f;
  float sx = ((float)xx + 0.5f) * ((float)WF / (float)WW) - 0.5f;
  sy = fminf(fmaxf(sy, 0.f), (float)(HF - 1));
  sx = fminf(fmaxf(sx, 0.f), (float)(WF - 1));
  int y0 = (int)sy, x0 = (int)sx;
  int y1 = min(y0 + 1, HF - 1), x1 = min(x0 + 1, WF - 1);
  float fy = sy - (float)y0, fx = sx - (float)x0;
  float w00 = (1.f - fy) * (1.f - fx), w01 = (1.f - fy) * fx;
  float w10 = fy * (1.f - fx),         w11 = fy * fx;
  const float* xb = x + (size_t)b * CC * HF * WF;
  __bf16* o = hin + (size_t)i * CP;
  for (int c = 0; c < CC; ++c) {
    const float* pc = xb + (size_t)c * HF * WF;
    float vv = w00 * pc[y0 * WF + x0] + w01 * pc[y0 * WF + x1] +
               w10 * pc[y1 * WF + x0] + w11 * pc[y1 * WF + x1];
    o[c] = (__bf16)vv;
  }
  float d = depth[i];
  d = (d != d) ? 0.f : d;                 // nan -> 0
  d = fminf(fmaxf(d, 0.f), 100.f);        // +inf -> 100, -inf -> 0, clip
  o[CC] = (__bf16)(d * 0.01f);
  for (int c = C1; c < CP; ++c) o[c] = (__bf16)0.f;  // zero K-padding
}

// ---------------------- pack w1 (OIHW f32) into per-lane WMMA-B bf16 layout --
// layout: [tap 0..8][kc 0..4][ntile 0..3][lane 0..31][j 0..15]
__global__ void pack_w1_kernel(const float* __restrict__ w1,
                               __bf16* __restrict__ wp) {
  int t = blockIdx.x * blockDim.x + threadIdx.x;
  if (t >= 9 * 5 * 4 * 32) return;
  int lane = t & 31;
  int q = t >> 5;
  int ntile = q & 3; q >>= 2;
  int kc = q % 5;
  int tap = q / 5;
  int n = ntile * 16 + (lane & 15);
  int half = lane >> 4;
  __bf16* o = wp + (size_t)t * 16;
  for (int j = 0; j < 16; ++j) {
    int k = kc * 32 + klocal_of(j, half);
    float v = 0.f;
    if (k < C1) v = w1[((size_t)n * C1 + k) * 9 + tap];
    o[j] = (__bf16)v;
  }
}

// ------------------------------------- conv1 3x3 (implicit GEMM, bf16 WMMA) --
// block = 8 waves; owns (b, y, 64-pixel x strip) x all 64 out channels.
// Stage 3 halo rows (66 px x 160 ch bf16) into LDS once via async global->LDS,
// then feed WMMA from ds_load_b128 (uniform addressing, no EXEC juggling,
// ~36x less L2 traffic than per-wave global fragment loads).
__global__ void __launch_bounds__(256) conv1_wmma_kernel(
    const __bf16* __restrict__ hin, const __bf16* __restrict__ wp,
    __bf16* __restrict__ hmid, float* __restrict__ stats) {
  __shared__ __bf16 tile[3 * TILE_PX * CP];   // 63360 B
  __shared__ float sg[NG * 2];

  int blk = blockIdx.x;                 // BB*HH*(WW/64) = 8192 blocks
  int xs64 = (blk & 7) * 64;
  int y = (blk >> 3) & (HH - 1);
  int b = blk >> 11;
  int tid = threadIdx.x;
  int lane = tid & 31, wave = tid >> 5;
  int ntile = wave & 3;
  int mt0 = wave >> 2;                  // 0 or 1; second tile is mt0+2
  int m = lane & 15, half = lane >> 4;

  if (tid < NG * 2) sg[tid] = 0.f;

  // ---- stage input tile: 3 rows x 66 px x 320 B = 3960 x 16-byte chunks ----
  const int NCHUNK = 3 * TILE_PX * (CP * 2 / 16);  // 3960
  for (int t = tid; t < NCHUNK; t += 256) {
    int c16 = t % (CP * 2 / 16);        // 16B chunk within pixel (0..19)
    int pr = t / (CP * 2 / 16);
    int p = pr % TILE_PX;               // local pixel 0..65
    int r = pr / TILE_PX;               // row 0..2
    int yy = y + r - 1;
    int xx = xs64 - 1 + p;
    __bf16* lp = &tile[(size_t)(r * TILE_PX + p) * CP + c16 * 8];
    if (((unsigned)yy < (unsigned)HH) && ((unsigned)xx < (unsigned)WW)) {
      const __bf16* gp = hin + ((size_t)(b * HH + yy) * WW + xx) * CP + c16 * 8;
#if __has_builtin(__builtin_amdgcn_global_load_async_to_lds_b128)
      __builtin_amdgcn_global_load_async_to_lds_b128(
          (AS1 vsi4*)gp, (AS3 vsi4*)lp, 0, 0);
#else
      *(uint4*)lp = *(const uint4*)gp;
#endif
    } else {
      *(uint4*)lp = make_uint4(0, 0, 0, 0);
    }
  }
#if __has_builtin(__builtin_amdgcn_global_load_async_to_lds_b128)
#if __has_builtin(__builtin_amdgcn_s_wait_asynccnt)
  __builtin_amdgcn_s_wait_asynccnt(0);
#else
  asm volatile("s_wait_asynccnt 0x0" ::: "memory");
#endif
#endif
  __syncthreads();

  v8f acc0 = {};
  v8f acc1 = {};

  for (int tap = 0; tap < 9; ++tap) {
    int dy = tap / 3 - 1, dx = tap % 3 - 1;
    int pl0 = mt0 * 16 + m + dx + 1;    // local pixel for mtile0 row m
    int pl1 = pl0 + 32;                 // mtile0+2
    const __bf16* r0 = &tile[(size_t)((dy + 1) * TILE_PX + pl0) * CP];
    const __bf16* r1 = &tile[(size_t)((dy + 1) * TILE_PX + pl1) * CP];
    for (int kc = 0; kc < 5; ++kc) {
      Frag16 bf;
      const __bf16* wb =
          wp + ((size_t)((tap * 5 + kc) * 4 + ntile) * 32 + lane) * 16;
      bf.q[0] = ((const uint4*)wb)[0];
      bf.q[1] = ((const uint4*)wb)[1];
      int cbase = kc * 32 + 8 * half;
      Frag16 a0, a1;
      a0.q[0] = *(const uint4*)(r0 + cbase);
      a0.q[1] = *(const uint4*)(r0 + cbase + 16);
      a1.q[0] = *(const uint4*)(r1 + cbase);
      a1.q[1] = *(const uint4*)(r1 + cbase + 16);
      acc0 = __builtin_amdgcn_wmma_f32_16x16x32_bf16(
          false, a0.v, false, bf.v, (short)0, acc0, false, false);
      acc1 = __builtin_amdgcn_wmma_f32_16x16x32_bf16(
          false, a1.v, false, bf.v, (short)0, acc1, false, false);
    }
  }

  // D layout: lane -> N = lane&15; VGPR v -> M = v + 8*(lane>>4)
  int co = ntile * 16 + m;
  float lsum = 0.f, lss = 0.f;
  {
    size_t rowpix = (size_t)(b * HH + y) * WW;
    int xbase = xs64 + mt0 * 16;
    for (int v = 0; v < 8; ++v) {
      int M = v + 8 * half;
      float val = acc0[v];
      hmid[(rowpix + xbase + M) * CO + co] = (__bf16)val;
      lsum += val; lss += val * val;
    }
    xbase = xs64 + (mt0 + 2) * 16;
    for (int v = 0; v < 8; ++v) {
      int M = v + 8 * half;
      float val = acc1[v];
      hmid[(rowpix + xbase + M) * CO + co] = (__bf16)val;
      lsum += val; lss += val * val;
    }
  }
  int g = co >> 2;
  atomicAdd(&sg[g * 2 + 0], lsum);
  atomicAdd(&sg[g * 2 + 1], lss);
  __syncthreads();
  if (tid < NG) {
    atomicAdd(&stats[(b * NG + tid) * 2 + 0], sg[tid * 2 + 0]);
    atomicAdd(&stats[(b * NG + tid) * 2 + 1], sg[tid * 2 + 1]);
  }
}

// ---------------------------------------------- groupnorm affine parameters --
__global__ void gn_param_kernel(const float* __restrict__ stats,
                                const float* __restrict__ gamma,
                                const float* __restrict__ beta,
                                float* __restrict__ scale,
                                float* __restrict__ bias) {
  int i = blockIdx.x * blockDim.x + threadIdx.x;  // BB*CO = 256
  if (i >= BB * CO) return;
  int b = i / CO, c = i % CO, g = c >> 2;
  float cnt = (float)((CO / NG) * HH * WW);
  float mu = stats[(b * NG + g) * 2 + 0] / cnt;
  float var = stats[(b * NG + g) * 2 + 1] / cnt - mu * mu;
  float inv = rsqrtf(var + 1e-5f);
  float s = gamma[c] * inv;
  scale[i] = s;
  bias[i] = beta[c] - mu * s;
}

// ------------------------- GN + SiLU + 1x1 conv -> logits, global max -------
__global__ void logw_kernel(const __bf16* __restrict__ hmid,
                            const float* __restrict__ scale,
                            const float* __restrict__ bias,
                            const float* __restrict__ w2,
                            const float* __restrict__ b2,
                            const float* __restrict__ logt,
                            const int* __restrict__ idx,
                            float* __restrict__ logw,
                            unsigned int* __restrict__ maxbits) {
  int i = blockIdx.x * blockDim.x + threadIdx.x;
  if (i >= NPIX) return;
  int b = i / (HH * WW);
  const __bf16* hp = hmid + (size_t)i * CO;
  const float* sc = scale + b * CO;
  const float* bi = bias + b * CO;
  float acc = 0.f;
  for (int c = 0; c < CO; ++c) {
    float t = (float)hp[c] * sc[c] + bi[c];
    float s = t / (1.f + __expf(-t));   // silu
    acc += s * w2[c];
  }
  float lw = (acc + b2[0]) * __expf(-logt[0]);
  logw[i] = lw;
  if (idx[i] >= 0) {
    unsigned u = __float_as_uint(lw);
    unsigned s2 = (u & 0x80000000u) ? ~u : (u | 0x80000000u);  // monotone map
    atomicMax(maxbits, s2);
  }
}

__device__ __forceinline__ float decode_max(unsigned s) {
  unsigned u = (s & 0x80000000u) ? (s ^ 0x80000000u) : ~s;
  return __uint_as_float(u);
}

// -------------------------------------------------- softmax weight sums ------
__global__ void wsum_kernel(const int* __restrict__ idx,
                            const float* __restrict__ logw,
                            const unsigned* __restrict__ maxbits,
                            float* __restrict__ wsum) {
  int i = blockIdx.x * blockDim.x + threadIdx.x;
  if (i >= NPIX) return;
  int id = idx[i];
  if (id < 0) return;
  float mx = decode_max(*maxbits);
  atomicAdd(&wsum[id], __expf(logw[i] - mx));
}

// ------------------------------------- weighted feature splat into BEV -------
__global__ void scatter_kernel(const int* __restrict__ idx,
                               const float* __restrict__ logw,
                               const unsigned* __restrict__ maxbits,
                               const float* __restrict__ wsum,
                               const __bf16* __restrict__ hin,
                               float* __restrict__ out) {
  int i = blockIdx.x * blockDim.x + threadIdx.x;
  if (i >= NPIX) return;
  int id = idx[i];
  if (id < 0) return;
  float mx = decode_max(*maxbits);
  float w = __expf(logw[i] - mx);
  float denom = fmaxf(wsum[id], 1e-4f);
  float s = w / denom;
  int b = id >> 16;
  int rem = id & 0xFFFF;
  int vx = rem >> 8, vy = rem & 255;
  // bev_emb[b, c, NX-1-vx, NY-1-vy] (transpose + double flip folded in)
  float* o = out + (((size_t)b * CC) * NXV + (NXV - 1 - vx)) * NYV + (NYV - 1 - vy);
  const __bf16* f = hin + (size_t)i * CP;
  for (int c = 0; c < CC; ++c) {
    atomicAdd(o + (size_t)c * NXV * NYV, (float)f[c] * s);
  }
}

// ------------------------------------------------------------- BEV mask ------
__global__ void mask_kernel(const float* __restrict__ wsum,
                            float* __restrict__ out) {
  int sId = blockIdx.x * blockDim.x + threadIdx.x;
  if (sId >= SVOX) return;
  int b = sId >> 16;
  int rem = sId & 0xFFFF;
  int vx = rem >> 8, vy = rem & 255;
  float m = (wsum[sId] > 1e-6f) ? 1.f : 0.f;
  out[(size_t)BB * CC * NXV * NYV +
      ((size_t)b * NXV + (NXV - 1 - vx)) * NYV + (NYV - 1 - vy)] = m;
}

// ============================================================================
extern "C" void kernel_launch(void* const* d_in, const int* in_sizes, int n_in,
                              void* d_out, int out_size, void* d_ws, size_t ws_size,
                              hipStream_t stream) {
  (void)in_sizes; (void)n_in; (void)ws_size;
  const float* x     = (const float*)d_in[0];
  const float* depth = (const float*)d_in[1];
  const float* Km    = (const float*)d_in[2];
  const float* c2e   = (const float*)d_in[3];
  const float* reso  = (const float*)d_in[4];
  const float* w1    = (const float*)d_in[5];
  const float* gamma = (const float*)d_in[6];
  const float* beta  = (const float*)d_in[7];
  const float* w2    = (const float*)d_in[8];
  const float* b2    = (const float*)d_in[9];
  const float* logt  = (const float*)d_in[10];
  float* out = (float*)d_out;

  char* p = (char*)d_ws;
  auto carve = [&](size_t bytes) {
    char* r = p;
    p += (bytes + 255) & ~(size_t)255;
    return r;
  };
  __bf16* hin   = (__bf16*)carve((size_t)NPIX * CP * 2);       // 160 MB
  __bf16* hmid  = (__bf16*)carve((size_t)NPIX * CO * 2);       //  64 MB
  __bf16* wp    = (__bf16*)carve((size_t)9 * 5 * 4 * 32 * 16 * 2);
  int*    idx   = (int*)   carve((size_t)NPIX * 4);
  float*  logw  = (float*) carve((size_t)NPIX * 4);
  float*  wsum  = (float*) carve((size_t)SVOX * 4);
  float*  stats = (float*) carve((size_t)BB * NG * 2 * 4);
  float*  gsc   = (float*) carve((size_t)BB * CO * 4);
  float*  gbi   = (float*) carve((size_t)BB * CO * 4);
  unsigned* mxb = (unsigned*)carve(256);

  (void)hipMemsetAsync(out, 0, (size_t)out_size * 4, stream);
  (void)hipMemsetAsync(wsum, 0, (size_t)SVOX * 4, stream);
  (void)hipMemsetAsync(stats, 0, (size_t)BB * NG * 2 * 4, stream);
  (void)hipMemsetAsync(mxb, 0, 4, stream);

  geom_kernel<<<NPIX / 256, 256, 0, stream>>>(depth, Km, c2e, reso, idx);
  build_hin_kernel<<<NPIX / 256, 256, 0, stream>>>(x, depth, hin);
  pack_w1_kernel<<<(9 * 5 * 4 * 32 + 255) / 256, 256, 0, stream>>>(w1, wp);
  conv1_wmma_kernel<<<BB * HH * (WW / 64), 256, 0, stream>>>(hin, wp, hmid, stats);
  gn_param_kernel<<<1, 256, 0, stream>>>(stats, gamma, beta, gsc, gbi);
  logw_kernel<<<NPIX / 256, 256, 0, stream>>>(hmid, gsc, gbi, w2, b2, logt, idx,
                                              logw, mxb);
  wsum_kernel<<<NPIX / 256, 256, 0, stream>>>(idx, logw, mxb, wsum);
  scatter_kernel<<<NPIX / 256, 256, 0, stream>>>(idx, logw, mxb, wsum, hin, out);
  mask_kernel<<<SVOX / 256, 256, 0, stream>>>(wsum, out);
}
